// DeepFusionLayers_66425964200554
// MI455X (gfx1250) — matrix operand order repaired and checked by reading, not compiled, not run on hardware
//
#include <hip/hip_runtime.h>
#include <hip/hip_bf16.h>

// ---------------------------------------------------------------------------
// Types for CDNA5 WMMA (gfx1250, wave32)
// ---------------------------------------------------------------------------
typedef __attribute__((ext_vector_type(16))) __bf16 bf16x16;
typedef __attribute__((ext_vector_type(2)))  __bf16 bf16x2;
typedef __attribute__((ext_vector_type(8)))  float  f32x8;

union Frag {
    bf16x16 v;
    float4  f4[2];
};

__device__ __forceinline__ unsigned short f32_to_bf16(float f) {
    unsigned int u = __float_as_uint(f);
    unsigned int r = (u + 0x7FFFu + ((u >> 16) & 1u)) >> 16;
    return (unsigned short)r;
}

__device__ __forceinline__ unsigned int pack_bf16x2(float lo, float hi) {
#if __has_builtin(__builtin_amdgcn_cvt_pk_bf16_f32)
    bf16x2 r = __builtin_amdgcn_cvt_pk_bf16_f32(lo, hi);
    return __builtin_bit_cast(unsigned int, r);
#else
    return (unsigned int)f32_to_bf16(lo) | ((unsigned int)f32_to_bf16(hi) << 16);
#endif
}

__device__ __forceinline__ void atomicMaxF32(float* addr, float val) {
    unsigned int* ai = (unsigned int*)addr;
    unsigned int old = *ai;
    while (__uint_as_float(old) < val) {
        unsigned int assumed = old;
        old = atomicCAS(ai, assumed, __float_as_uint(val));
        if (old == assumed) break;
    }
}

// ---------------------------------------------------------------------------
// Implicit-GEMM conv (KS=1 or KS=3 pad1), bf16 WMMA, f32 accumulate.
// Template <KS,H,W,NSPLIT>: all index math is mul/shift by constants; the
// 3-way channel-split pointer select only exists for the concat (CAB) conv.
// Weights are packed [Cout][Kpad] bf16 with zero tail (Kpad = ceil64(K)) so
// the K loop runs in chunks of 64.
// Block = 256 threads = 8 waves; tile = 48 Cout (MT=3 sub-tiles) x 256 pixels.
// Each wave owns MT x 2 accumulators; staged B chunk reused by 3 A fragments
// -> 12 WMMA per staging round and 3x less im2col gather traffic.
// Optional fused 8x8 max-pool epilogue (atomic max into low-res buffer).
// ---------------------------------------------------------------------------
#define CTN  256   // pixels per block (== blockDim)
#define LSTR 72    // LDS row stride in bf16 elems (144B: 16B aligned, bank-spread)
#define MT   3     // cout sub-tiles per block (48 cout)

template <int KS, int H, int W, int NSPLIT>
__global__ __launch_bounds__(256)
void k_conv_gemm(const float* __restrict__ in0, const float* __restrict__ in1,
                 const float* __restrict__ in2, int split1, int split2,
                 const unsigned short* __restrict__ Wp, const float* __restrict__ bias,
                 float* __restrict__ out, float* __restrict__ out_pool,
                 int Cin, int Cout, int Npix, int KP)
{
    constexpr int KK  = KS * KS;
    constexpr int HW  = H * W;
    constexpr int PAD = (KS == 3) ? 1 : 0;
    __shared__ unsigned short lds[CTN * LSTR];
    unsigned int* lds32 = (unsigned int*)lds;

    const int tid  = threadIdx.x;
    const int lane = tid & 31, wave = tid >> 5;
    const int half = lane >> 4, nl = lane & 15;
    const int pixBlock = blockIdx.x * CTN;
    const int cout0    = blockIdx.y * (16 * MT);

    const f32x8 fz = {0.f, 0.f, 0.f, 0.f, 0.f, 0.f, 0.f, 0.f};
    f32x8 acc[MT][2];
#pragma unroll
    for (int m = 0; m < MT; ++m) { acc[m][0] = fz; acc[m][1] = fz; }

    // ---- per-thread staging pixel (loop-invariant): pl == tid ----
    int spix = pixBlock + tid;
    spix = spix < Npix ? spix : Npix - 1;
    const int sb  = spix / HW;
    const int sre = spix - sb * HW;
    const int sy  = sre / W, sx = sre - sy * W;

    // A rows are contiguous in K: lane nl -> row (cout0 + m*16 + nl), K base half*8
    const unsigned short* wrow[MT];
#pragma unroll
    for (int m = 0; m < MT; ++m)
        wrow[m] = Wp + (size_t)(cout0 + m * 16 + nl) * KP + half * 8;

    for (int kc = 0; kc < KP; kc += 64) {
        __syncthreads();
        // stage im2col patches: 32 K-pairs for this thread's pixel, packed bf16x2
#pragma unroll 8
        for (int kp = 0; kp < 32; ++kp) {
            float vv[2];
#pragma unroll
            for (int uu = 0; uu < 2; ++uu) {
                int k    = kc + kp * 2 + uu;
                int cin0 = k / KK;                       // wave-uniform -> SALU
                int t    = k - cin0 * KK;
                int cin  = cin0 < Cin ? cin0 : Cin - 1;  // tail-safe (A tail is zero)
                const float* src; int c, Cg;
                if (NSPLIT == 1) { src = in0; c = cin; Cg = Cin; }
                else {
                    if (cin < split1)      { src = in0; c = cin;          Cg = split1; }
                    else if (cin < split2) { src = in1; c = cin - split1; Cg = split2 - split1; }
                    else                   { src = in2; c = cin - split2; Cg = Cin - split2; }
                }
                float v;
                if (KS == 1) {
                    v = src[((size_t)(sb * Cg + c)) * HW + sre];
                } else {
                    int ky = t / KS - PAD, kx = t % KS - PAD;
                    int yy = sy + ky, xx = sx + kx;
                    bool inb = ((unsigned)yy < (unsigned)H) && ((unsigned)xx < (unsigned)W);
                    int yyc = inb ? yy : sy;             // safe address, no branch
                    int xxc = inb ? xx : sx;
                    v = src[((size_t)(sb * Cg + c)) * HW + yyc * W + xxc];
                    v = inb ? v : 0.0f;
                }
                vv[uu] = v;
            }
            lds32[tid * (LSTR / 2) + kp] = pack_bf16x2(vv[0], vv[1]);
        }
        __syncthreads();

        // B fragments (shared across the MT cout sub-tiles)
        Frag b0a, b0b, b1a, b1b;
        const unsigned short* brow = lds + (wave * 32 + nl) * LSTR + half * 8;
        b0a.f4[0] = *(const float4*)(brow);                  // pixels wave*32 + nl
        b0a.f4[1] = *(const float4*)(brow + 16);
        b0b.f4[0] = *(const float4*)(brow + 32);
        b0b.f4[1] = *(const float4*)(brow + 48);
        b1a.f4[0] = *(const float4*)(brow + 16 * LSTR);      // pixels wave*32+16+nl
        b1a.f4[1] = *(const float4*)(brow + 16 * LSTR + 16);
        b1b.f4[0] = *(const float4*)(brow + 16 * LSTR + 32);
        b1b.f4[1] = *(const float4*)(brow + 16 * LSTR + 48);

#pragma unroll
        for (int m = 0; m < MT; ++m) {
            Frag fa0, fa1;
            fa0.f4[0] = *(const float4*)(wrow[m] + kc);       // K = kc+half*8     ..+7
            fa0.f4[1] = *(const float4*)(wrow[m] + kc + 16);  // K = kc+16+half*8  ..+7
            fa1.f4[0] = *(const float4*)(wrow[m] + kc + 32);
            fa1.f4[1] = *(const float4*)(wrow[m] + kc + 48);
            acc[m][0] = __builtin_amdgcn_wmma_f32_16x16x32_bf16(false, fa0.v, false, b0a.v,
                                                                (short)0, acc[m][0], false, false);
            acc[m][0] = __builtin_amdgcn_wmma_f32_16x16x32_bf16(false, fa1.v, false, b0b.v,
                                                                (short)0, acc[m][0], false, false);
            acc[m][1] = __builtin_amdgcn_wmma_f32_16x16x32_bf16(false, fa0.v, false, b1a.v,
                                                                (short)0, acc[m][1], false, false);
            acc[m][1] = __builtin_amdgcn_wmma_f32_16x16x32_bf16(false, fa1.v, false, b1b.v,
                                                                (short)0, acc[m][1], false, false);
        }
    }

    // epilogue: C/D layout -> cout = coutBase + half*8 + r, pixel = tile base + nl
    auto emit = [&](int pix, int coutBase, const f32x8& acc_) {
        if (pix >= Npix) return;
        int b = pix / HW; int rem = pix - b * HW;
        int y = rem / W,  x  = rem - y * W;
#pragma unroll
        for (int r = 0; r < 8; r++) {
            int cout  = coutBase + half * 8 + r;
            float val = acc_[r] + (bias ? bias[cout] : 0.0f);
            if (out_pool) {
                constexpr int Hs = H >> 3, Ws = W >> 3;
                atomicMaxF32(out_pool + ((size_t)(b * Cout + cout) * Hs + (y >> 3)) * Ws + (x >> 3), val);
            } else {
                out[((size_t)(b * Cout + cout)) * HW + (size_t)y * W + x] = val;
            }
        }
    };
#pragma unroll
    for (int m = 0; m < MT; ++m) {
        emit(pixBlock + wave * 32 + nl,      cout0 + m * 16, acc[m][0]);
        emit(pixBlock + wave * 32 + 16 + nl, cout0 + m * 16, acc[m][1]);
    }
}

// ---------------------------------------------------------------------------
// Weight preparation (packed [Cout][Kpad] bf16, zero tail)
// ---------------------------------------------------------------------------
__global__ void k_prep_mkw(const float* __restrict__ w,  const float* __restrict__ wc,
                           const float* __restrict__ wh, const float* __restrict__ wv,
                           const float* __restrict__ wa, const float* __restrict__ s4,
                           unsigned short* __restrict__ weff)   // [96][896]
{
    int e = blockIdx.x * 256 + threadIdx.x;
    if (e >= 96 * 896) return;
    int o = e / 896, k = e - o * 896;
    if (k >= 864) { weff[e] = 0; return; }
    int i = k / 9, t = k - i * 9;
    int oi = o * 96 + i, src = oi * 9 + t;
    float s0 = s4[0], s1 = s4[1], s2 = s4[2], s3 = s4[3];
    float wcd = wc[src];
    if (t == 4) { float sum = 0.f; for (int tt = 0; tt < 9; tt++) sum += wc[oi * 9 + tt]; wcd -= sum; }
    float whd = 0.f;
    if (t == 0 || t == 3 || t == 6)      whd =  wh[oi * 3 + t / 3];
    else if (t == 2 || t == 5 || t == 8) whd = -wh[oi * 3 + (t - 2) / 3];
    float wvd = 0.f;
    if (t < 3)       wvd =  wv[oi * 3 + t];
    else if (t >= 6) wvd = -wv[oi * 3 + t - 6];
    const int PERM[9] = {3, 0, 1, 6, 4, 2, 7, 8, 5};
    float wad = wa[src] - wa[oi * 9 + PERM[t]];
    weff[e] = f32_to_bf16(w[src] + s0 * wcd + s1 * whd + s2 * wvd + s3 * wad);
}

__global__ void k_prep_bias(const float* b, const float* bc, const float* bh,
                            const float* bv, const float* ba, const float* s4,
                            float* beff)
{
    int i = threadIdx.x;
    if (i < 96) beff[i] = b[i] + s4[0] * bc[i] + s4[1] * bh[i] + s4[2] * bv[i] + s4[3] * ba[i];
}

__global__ void k_cvt_pad(const float* __restrict__ in, unsigned short* __restrict__ out,
                          int rows, int Kin, int Kp)
{
    int e = blockIdx.x * 256 + threadIdx.x;
    if (e >= rows * Kp) return;
    int r = e / Kp, k = e - r * Kp;
    out[e] = (k < Kin) ? f32_to_bf16(in[r * Kin + k]) : (unsigned short)0;
}

__global__ void k_fill(float* p, float v, int n)
{
    int i = blockIdx.x * 256 + threadIdx.x;
    if (i < n) p[i] = v;
}

// ---------------------------------------------------------------------------
// DTKSA small ops (all at 24x24 resolution)
// ---------------------------------------------------------------------------
__global__ void k_dwconv(const float* __restrict__ in, const float* __restrict__ w,
                         const float* __restrict__ bias, float* __restrict__ out)
{
    int i = blockIdx.x * 256 + threadIdx.x;           // 2*288*24*24
    if (i >= 2 * 288 * 24 * 24) return;
    int x = i % 24, y = (i / 24) % 24, c = (i / 576) % 288, b = i / (576 * 288);
    const float* wp = w + c * 9;
    float s = bias[c];
    for (int ky = -1; ky <= 1; ky++)
        for (int kx = -1; kx <= 1; kx++) {
            int yy = y + ky, xx = x + kx;
            if ((unsigned)yy < 24u && (unsigned)xx < 24u)
                s += wp[(ky + 1) * 3 + (kx + 1)] * in[((b * 288 + c) * 24 + yy) * 24 + xx];
        }
    out[i] = s;
}

__global__ void k_qknorm(float* __restrict__ qkvdw)
{
    int row = blockIdx.x;                 // 0..383: (b, ch<192)
    int b = row / 192, ch = row % 192;
    float* p = qkvdw + (b * 288 + ch) * 576;
    __shared__ float red[64];
    float s = 0.f;
    for (int i = threadIdx.x; i < 576; i += 64) { float v = p[i]; s += v * v; }
    red[threadIdx.x] = s; __syncthreads();
    for (int off = 32; off > 0; off >>= 1) {
        if ((int)threadIdx.x < off) red[threadIdx.x] += red[threadIdx.x + off];
        __syncthreads();
    }
    float inv = 1.0f / fmaxf(sqrtf(red[0]), 1e-12f);
    for (int i = threadIdx.x; i < 576; i += 64) p[i] *= inv;
}

__global__ void k_attn(const float* __restrict__ qkvdw, const float* __restrict__ temp,
                       float* __restrict__ attn)
{
    int bh = blockIdx.x;                  // 0..3
    int b = bh >> 1, h = bh & 1;
    float t = temp[h];
    for (int e = threadIdx.x; e < 2304; e += blockDim.x) {
        int c = e / 48, d = e % 48;
        const float* q = qkvdw + (b * 288 + h * 48 + c) * 576;
        const float* k = qkvdw + (b * 288 + 96 + h * 48 + d) * 576;
        float s = 0.f;
        for (int n = 0; n < 576; n++) s += q[n] * k[n];
        attn[bh * 2304 + e] = s * t;
    }
}

// per-row: sort copy, 4 top-k thresholds, combined coefficient matrix
__global__ void k_attn_mask(const float* __restrict__ attn, const float* __restrict__ aw,
                            float* __restrict__ M)
{
    int row = blockIdx.x * blockDim.x + threadIdx.x;   // 0..191
    if (row >= 192) return;
    const float* a = attn + row * 48;
    float s[48], o[48];
    for (int i = 0; i < 48; i++) { s[i] = a[i]; o[i] = 0.f; }
    for (int i = 1; i < 48; i++) {                     // insertion sort, descending
        float key = s[i]; int j = i - 1;
        while (j >= 0 && s[j] < key) { s[j + 1] = s[j]; j--; }
        s[j + 1] = key;
    }
    const int kks[4] = {24, 32, 36, 38};               // C/2, 2C/3, 3C/4, 4C/5 for C=48
    float maxv = s[0];
    for (int j = 0; j < 4; j++) {
        float th = s[kks[j] - 1];
        float sum = 0.f;
        for (int d = 0; d < 48; d++) if (a[d] >= th) sum += expf(a[d] - maxv);
        float wj = aw[j] / sum;
        for (int d = 0; d < 48; d++) if (a[d] >= th) o[d] += wj * expf(a[d] - maxv);
    }
    float* m = M + row * 48;
    for (int d = 0; d < 48; d++) m[d] = o[d];
}

__global__ void k_av_gelu(const float* __restrict__ M, const float* __restrict__ qkvdw,
                          float* __restrict__ out)
{
    int idx = blockIdx.x * 256 + threadIdx.x;          // 2*96*576
    if (idx >= 2 * 96 * 576) return;
    int n = idx % 576, ch = (idx / 576) % 96, b = idx / (576 * 96);
    int h = ch / 48, c = ch % 48;
    const float* m = M + ((b * 2 + h) * 48 + c) * 48;
    const float* v = qkvdw + (b * 288 + 192 + h * 48) * 576 + n;
    float s = 0.f;
    for (int d = 0; d < 48; d++) s += m[d] * v[d * 576];
    out[idx] = 0.5f * s * (1.0f + erff(s * 0.70710678118654752f));   // exact GELU
}

__global__ void k_upsample8(const float* __restrict__ in, float* __restrict__ out)
{
    int i = blockIdx.x * 256 + threadIdx.x;            // 2*96*192*192
    if (i >= 2 * 96 * 192 * 192) return;
    int x = i % 192; int r = i / 192; int y = r % 192; int bc = r / 192;
    out[i] = in[(bc * 24 + (y >> 3)) * 24 + (x >> 3)];
}

// ---------------------------------------------------------------------------
// EAF: BN statistics + sigmoid blend (in place on x)
// ---------------------------------------------------------------------------
__global__ void k_bnstats(const float* __restrict__ xin, float* __restrict__ mean_out,
                          float* __restrict__ inv_out)
{
    int c = blockIdx.x;                                // 0..47
    const int HW = 192 * 192;
    __shared__ float r1[256], r2[256];
    float s = 0.f, s2 = 0.f;
    for (int b = 0; b < 2; b++) {
        const float* p = xin + (size_t)(b * 48 + c) * HW;
        for (int i = threadIdx.x; i < HW; i += 256) { float v = p[i]; s += v; s2 += v * v; }
    }
    r1[threadIdx.x] = s; r2[threadIdx.x] = s2; __syncthreads();
    for (int off = 128; off > 0; off >>= 1) {
        if ((int)threadIdx.x < off) { r1[threadIdx.x] += r1[threadIdx.x + off]; r2[threadIdx.x] += r2[threadIdx.x + off]; }
        __syncthreads();
    }
    if (threadIdx.x == 0) {
        float N = 2.0f * HW;
        float m = r1[0] / N;
        float var = r2[0] / N - m * m;
        mean_out[c] = m;
        inv_out[c]  = rsqrtf(var + 1e-5f);
    }
}

__global__ void k_eaf_apply(float* __restrict__ xbuf, const float* __restrict__ ybuf,
                            const float* __restrict__ xk, const float* __restrict__ yq,
                            const float* __restrict__ stats)
{
    int p = blockIdx.x * 256 + threadIdx.x;            // 2*192*192
    const int HW = 192 * 192;
    if (p >= 2 * HW) return;
    int b = p / HW, hw = p % HW;
    float s = 0.f;
    for (int c = 0; c < 48; c++) {
        float xv = (xk[(size_t)(b * 48 + c) * HW + hw] - stats[c])      * stats[48 + c];
        float yv = (yq[(size_t)(b * 48 + c) * HW + hw] - stats[96 + c]) * stats[144 + c];
        s += xv * yv;
    }
    float sim = 1.0f / (1.0f + expf(-s));
    for (int c = 0; c < 96; c++) {
        size_t idx = (size_t)(b * 96 + c) * HW + hw;
        float xv = xbuf[idx], yv = ybuf[idx];
        xbuf[idx] = sim * xv + (1.0f - sim) * yv;
    }
}

// ---------------------------------------------------------------------------
// Host orchestration
// ---------------------------------------------------------------------------
extern "C" void kernel_launch(void* const* d_in, const int* in_sizes, int n_in,
                              void* d_out, int out_size, void* d_ws, size_t ws_size,
                              hipStream_t stream)
{
    const float* X    = (const float*)d_in[0];
    const float* HFp  = (const float*)d_in[1];
    const float* LFp  = (const float*)d_in[2];
    const float* mkw_w  = (const float*)d_in[3];
    const float* mkw_b  = (const float*)d_in[4];
    const float* mkw_wc = (const float*)d_in[5];
    const float* mkw_bc = (const float*)d_in[6];
    const float* mkw_wh = (const float*)d_in[7];
    const float* mkw_bh = (const float*)d_in[8];
    const float* mkw_wv = (const float*)d_in[9];
    const float* mkw_bv = (const float*)d_in[10];
    const float* mkw_wa = (const float*)d_in[11];
    const float* mkw_ba = (const float*)d_in[12];
    const float* mkw_sc = (const float*)d_in[13];
    const float* dt_temp  = (const float*)d_in[14];
    const float* dt_wqkv  = (const float*)d_in[15];
    const float* dt_bqkv  = (const float*)d_in[16];
    const float* dt_wdw   = (const float*)d_in[17];
    const float* dt_bdw   = (const float*)d_in[18];
    const float* dt_wproj = (const float*)d_in[19];
    const float* dt_bproj = (const float*)d_in[20];
    const float* dt_attnw = (const float*)d_in[21];
    const float* eafh_wx = (const float*)d_in[22];
    const float* eafh_wy = (const float*)d_in[23];
    const float* eafl_wx = (const float*)d_in[24];
    const float* eafl_wy = (const float*)d_in[25];
    const float* cab_w = (const float*)d_in[26];
    const float* cab_b = (const float*)d_in[27];

    const long FULL   = 2L * 96 * 192 * 192;    // 7,077,888
    const long HALF   = FULL / 2;
    const long LOW288 = 2L * 288 * 24 * 24;     // 331,776
    const long ATT    = 4L * 48 * 48;           // 9,216
    const long AV     = 2L * 96 * 24 * 24;      // 110,592
    const int  NPIX   = 2 * 192 * 192;          // 73,728
    const int  NLOW   = 2 * 24 * 24;            // 1,152

    float* ws   = (float*)d_ws;
    float* x1b  = ws;
    float* x2b  = ws + FULL;
    float* t0   = ws + 2 * FULL;
    float* xkb  = ws + 3 * FULL;
    float* yqb  = ws + 3 * FULL + HALF;
    float* qkv    = ws + 4 * FULL;
    float* qkvdw  = qkv + LOW288;
    float* attn   = qkvdw + LOW288;
    float* Mb     = attn + ATT;
    float* avout  = Mb + ATT;
    float* projo  = avout + AV;
    float* stats  = projo + AV;                  // 192 used
    float* beff   = stats + 256;                 // 2*96 used
    // packed bf16 weights, [rows][Kpad], Kpad multiple of 64
    unsigned short* bfb   = (unsigned short*)(beff + 256);
    unsigned short* mkwW  = bfb;                 // 2 * 96*896  = 172032
    unsigned short* qkvW  = mkwW + 2 * 96 * 896; // 2 * 288*128 = 73728
    unsigned short* projW = qkvW + 2 * 288 * 128;// 2 * 96*128  = 24576
    unsigned short* ehxW  = projW + 2 * 96 * 128;// 2 * 48*128  = 12288 each
    unsigned short* ehyW  = ehxW + 2 * 48 * 128;
    unsigned short* elxW  = ehyW + 2 * 48 * 128;
    unsigned short* elyW  = elxW + 2 * 48 * 128;
    unsigned short* cabW  = elyW + 2 * 48 * 128; // 96*2624 = 251904

    // ---- weight prep (both layers) ----
    for (int l = 0; l < 2; l++) {
        k_prep_mkw<<<(96 * 896 + 255) / 256, 256, 0, stream>>>(
            mkw_w + l * 82944, mkw_wc + l * 82944, mkw_wh + l * 27648,
            mkw_wv + l * 27648, mkw_wa + l * 82944, mkw_sc + l * 4, mkwW + l * 96 * 896);
        k_prep_bias<<<1, 96, 0, stream>>>(
            mkw_b + l * 96, mkw_bc + l * 96, mkw_bh + l * 96,
            mkw_bv + l * 96, mkw_ba + l * 96, mkw_sc + l * 4, beff + l * 96);
        k_cvt_pad<<<(288 * 128 + 255) / 256, 256, 0, stream>>>(dt_wqkv + l * 27648, qkvW + l * 288 * 128, 288, 96, 128);
        k_cvt_pad<<<(96 * 128 + 255) / 256, 256, 0, stream>>>(dt_wproj + l * 9216, projW + l * 96 * 128, 96, 96, 128);
        k_cvt_pad<<<(48 * 128 + 255) / 256, 256, 0, stream>>>(eafh_wx + l * 4608, ehxW + l * 48 * 128, 48, 96, 128);
        k_cvt_pad<<<(48 * 128 + 255) / 256, 256, 0, stream>>>(eafh_wy + l * 4608, ehyW + l * 48 * 128, 48, 96, 128);
        k_cvt_pad<<<(48 * 128 + 255) / 256, 256, 0, stream>>>(eafl_wx + l * 4608, elxW + l * 48 * 128, 48, 96, 128);
        k_cvt_pad<<<(48 * 128 + 255) / 256, 256, 0, stream>>>(eafl_wy + l * 4608, elyW + l * 48 * 128, 48, 96, 128);
    }
    k_cvt_pad<<<(96 * 2624 + 255) / 256, 256, 0, stream>>>(cab_w, cabW, 96, 2592, 2624);

    const int GFULL = (NPIX + CTN - 1) / CTN;    // 288
    const int GLOW  = (NLOW + CTN - 1) / CTN;    // 5

    // ---- main pipeline ----
    const float* xin = X;
    float* feats[2] = {x1b, x2b};
    for (int l = 0; l < 2; l++) {
        // MKW 3x3 conv (WMMA): xin -> t0
        {
            dim3 g(GFULL, 96 / 48);
            k_conv_gemm<3, 192, 192, 1><<<g, 256, 0, stream>>>(xin, xin, xin, 96, 96,
                mkwW + l * 96 * 896, beff + l * 96, t0, nullptr, 96, 96, NPIX, 896);
        }
        // qkv 1x1 conv (WMMA) with fused 8x8 maxpool -> qkv (low res)
        k_fill<<<((int)LOW288 + 255) / 256, 256, 0, stream>>>(qkv, -3.0e38f, (int)LOW288);
        {
            dim3 g(GFULL, 288 / 48);
            k_conv_gemm<1, 192, 192, 1><<<g, 256, 0, stream>>>(t0, t0, t0, 96, 96,
                qkvW + l * 288 * 128, dt_bqkv + l * 288, nullptr, qkv, 96, 288, NPIX, 128);
        }
        // depthwise 3x3 at 24x24
        k_dwconv<<<((int)LOW288 + 255) / 256, 256, 0, stream>>>(qkv, dt_wdw + l * 288 * 9, dt_bdw + l * 288, qkvdw);
        // q/k row normalization
        k_qknorm<<<384, 64, 0, stream>>>(qkvdw);
        // attention scores, top-k masked combined softmax, M@V + GELU
        k_attn<<<4, 256, 0, stream>>>(qkvdw, dt_temp + l * 2, attn);
        k_attn_mask<<<1, 192, 0, stream>>>(attn, dt_attnw + l * 4, Mb);
        k_av_gelu<<<((int)AV + 255) / 256, 256, 0, stream>>>(Mb, qkvdw, avout);
        // proj 1x1 conv at low res (WMMA)
        {
            dim3 g(GLOW, 96 / 48);
            k_conv_gemm<1, 24, 24, 1><<<g, 256, 0, stream>>>(avout, avout, avout, 96, 96,
                projW + l * 96 * 128, dt_bproj + l * 96, projo, nullptr, 96, 96, NLOW, 128);
        }
        // nearest x8 upsample -> feats[l]
        float* xf = feats[l];
        k_upsample8<<<((int)FULL + 255) / 256, 256, 0, stream>>>(projo, xf);

        // EAF with hf
        {
            dim3 g(GFULL, 48 / 48);
            k_conv_gemm<1, 192, 192, 1><<<g, 256, 0, stream>>>(xf, xf, xf, 96, 96,
                ehxW + l * 48 * 128, nullptr, xkb, nullptr, 96, 48, NPIX, 128);
            k_conv_gemm<1, 192, 192, 1><<<g, 256, 0, stream>>>(HFp, HFp, HFp, 96, 96,
                ehyW + l * 48 * 128, nullptr, yqb, nullptr, 96, 48, NPIX, 128);
        }
        k_bnstats<<<48, 256, 0, stream>>>(xkb, stats + 0,  stats + 48);
        k_bnstats<<<48, 256, 0, stream>>>(yqb, stats + 96, stats + 144);
        k_eaf_apply<<<(NPIX + 255) / 256, 256, 0, stream>>>(xf, HFp, xkb, yqb, stats);

        // EAF with lf
        {
            dim3 g(GFULL, 48 / 48);
            k_conv_gemm<1, 192, 192, 1><<<g, 256, 0, stream>>>(xf, xf, xf, 96, 96,
                elxW + l * 48 * 128, nullptr, xkb, nullptr, 96, 48, NPIX, 128);
            k_conv_gemm<1, 192, 192, 1><<<g, 256, 0, stream>>>(LFp, LFp, LFp, 96, 96,
                elyW + l * 48 * 128, nullptr, yqb, nullptr, 96, 48, NPIX, 128);
        }
        k_bnstats<<<48, 256, 0, stream>>>(xkb, stats + 0,  stats + 48);
        k_bnstats<<<48, 256, 0, stream>>>(yqb, stats + 96, stats + 144);
        k_eaf_apply<<<(NPIX + 255) / 256, 256, 0, stream>>>(xf, LFp, xkb, yqb, stats);

        xin = xf;
    }

    // CAB: 3x3 conv over concat(x, feats1, feats2) -> d_out (WMMA)
    {
        dim3 g(GFULL, 96 / 48);
        k_conv_gemm<3, 192, 192, 3><<<g, 256, 0, stream>>>(X, x1b, x2b, 96, 192,
            cabW, cab_b, (float*)d_out, nullptr, 288, 96, NPIX, 2624);
    }
}
